// Head_31722628448420
// MI455X (gfx1250) — matrix operand order repaired and checked by reading, not compiled, not run on hardware
//
#include <hip/hip_runtime.h>

// Problem constants (match reference): B=8, S=2048, E=1024, D=64
#define BB 8
#define SS 2048
#define EE 1024
#define DD 64
#define BSR (BB * SS) // 16384 total rows

// Workspace layout (bf16, total ~6.4 MB):
//   q  : [BSR][D]     row-major
//   k  : [BSR][D]     row-major
//   vT : [D][BSR]     transposed (keys contiguous -> contiguous PV B-frags)
//   Wt : [3][D][E]    transposed bf16 weights (B-frag K-elements contiguous)

typedef __attribute__((ext_vector_type(16))) __bf16 v16bf;
typedef __attribute__((ext_vector_type(8)))  float  v8f;

union Frag16 {
    v16bf v;
    uint4 q[2];
};

__device__ __forceinline__ __bf16 to_bf16(float f) { return (__bf16)f; }

__device__ __forceinline__ v8f wmma_bf16(const v16bf& a, const v16bf& b, v8f c) {
    // emits v_wmma_f32_16x16x32_bf16
    return __builtin_amdgcn_wmma_f32_16x16x32_bf16(
        /*neg_a=*/false, a, /*neg_b=*/false, b,
        /*c_mod=*/(short)0, c, /*reuse_a=*/false, /*reuse_b=*/false);
}

// ---------------------------------------------------------------------------
// Kernel 0: one-time weight transpose + f32 -> bf16.
//   Wt[w][d][e] = bf16(W_w[e][d]);  3*D*E = 196608 elements.
// ---------------------------------------------------------------------------
__global__ __launch_bounds__(256)
void wcvt_kernel(const float* __restrict__ Wq,
                 const float* __restrict__ Wk,
                 const float* __restrict__ Wv,
                 __bf16* __restrict__ wt)
{
    int idx = blockIdx.x * 256 + threadIdx.x;   // 0 .. 3*D*E-1
    int w   = idx / (DD * EE);
    int rem = idx - w * (DD * EE);
    int d   = rem / EE;
    int e   = rem - d * EE;
    const float* W = (w == 0) ? Wq : ((w == 1) ? Wk : Wv);
    wt[idx] = to_bf16(W[(size_t)e * DD + d]);
}

// ---------------------------------------------------------------------------
// Kernel 1: fused QKV projection.
// grid = BSR/16 blocks, 96 threads (3 waves: wave0->q, wave1->k, wave2->vT).
// x tile (16x32) staged once per K-step in LDS as bf16 in A-frag order.
// B-fragments load straight from bf16 Wt (contiguous b128 pairs); all four
// are materialized before the WMMA chain so loads clause together and the
// XDL pipe is not gated on loadcnt==0 per fragment.
// ---------------------------------------------------------------------------
__global__ __launch_bounds__(96)
void qkv_proj_kernel(const float* __restrict__ x,
                     const __bf16* __restrict__ wt,     // [3][D][E]
                     __bf16* __restrict__ qout,         // [BSR][D]
                     __bf16* __restrict__ kout,         // [BSR][D]
                     __bf16* __restrict__ vtout)        // [D][BSR]
{
    // 16 rows x 32 cols bf16, row stride 72 (=36 dwords: conflict-free
    // across 16 lanes, rows 16B-aligned for b128 access)
    __shared__ __align__(16) __bf16 xl[16 * 72];

    const int tid   = threadIdx.x;
    const int wave  = tid >> 5;
    const int lane  = tid & 31;
    const int lrow  = lane & 15;
    const int lhalf = lane >> 4;
    const int koff  = 8 * lhalf;

    const int rowbase = blockIdx.x * 16;
    const __bf16* Wt = wt + (size_t)wave * DD * EE;

    v8f acc[4];
    acc[0] = v8f{}; acc[1] = v8f{}; acc[2] = v8f{}; acc[3] = v8f{};

    for (int k0 = 0; k0 < EE; k0 += 32) {
        // ---- stage x tile: 64 threads convert 8 floats each -> bf16 LDS
        if (tid < 64) {
            int r  = tid >> 2;
            int cg = (tid & 3) * 8;
            const float4* src =
                reinterpret_cast<const float4*>(&x[(size_t)(rowbase + r) * EE + k0 + cg]);
            float4 a = src[0];
            float4 b = src[1];
            union { __bf16 h[8]; uint4 q; } pk;
            pk.h[0] = to_bf16(a.x); pk.h[1] = to_bf16(a.y);
            pk.h[2] = to_bf16(a.z); pk.h[3] = to_bf16(a.w);
            pk.h[4] = to_bf16(b.x); pk.h[5] = to_bf16(b.y);
            pk.h[6] = to_bf16(b.z); pk.h[7] = to_bf16(b.w);
            *reinterpret_cast<uint4*>(&xl[r * 72 + cg]) = pk.q;
        }
        __syncthreads();

        // ---- all four B fragments first (one load clause, 8x b128)
        Frag16 b0, b1, b2, b3;
        {
            const __bf16* wp0 = &Wt[(size_t)(0 * 16 + lrow) * EE + k0 + 16 * lhalf];
            const __bf16* wp1 = &Wt[(size_t)(1 * 16 + lrow) * EE + k0 + 16 * lhalf];
            const __bf16* wp2 = &Wt[(size_t)(2 * 16 + lrow) * EE + k0 + 16 * lhalf];
            const __bf16* wp3 = &Wt[(size_t)(3 * 16 + lrow) * EE + k0 + 16 * lhalf];
            b0.q[0] = *reinterpret_cast<const uint4*>(&wp0[0]);
            b0.q[1] = *reinterpret_cast<const uint4*>(&wp0[8]);
            b1.q[0] = *reinterpret_cast<const uint4*>(&wp1[0]);
            b1.q[1] = *reinterpret_cast<const uint4*>(&wp1[8]);
            b2.q[0] = *reinterpret_cast<const uint4*>(&wp2[0]);
            b2.q[1] = *reinterpret_cast<const uint4*>(&wp2[8]);
            b3.q[0] = *reinterpret_cast<const uint4*>(&wp3[0]);
            b3.q[1] = *reinterpret_cast<const uint4*>(&wp3[8]);
        }

        // ---- A fragment: two b128 LDS loads (layout already A-frag order)
        Frag16 a;
        a.q[0] = *reinterpret_cast<const uint4*>(&xl[lrow * 72 + 2 * koff]);
        a.q[1] = *reinterpret_cast<const uint4*>(&xl[lrow * 72 + 8 + 2 * koff]);

        // ---- back-to-back WMMA chain
        acc[0] = wmma_bf16(a.v, b0.v, acc[0]);
        acc[1] = wmma_bf16(a.v, b1.v, acc[1]);
        acc[2] = wmma_bf16(a.v, b2.v, acc[2]);
        acc[3] = wmma_bf16(a.v, b3.v, acc[3]);

        __syncthreads(); // xl reused next iteration
    }

    // ---- epilogue: C layout -> bf16 workspace
#pragma unroll
    for (int nt = 0; nt < 4; ++nt) {
#pragma unroll
        for (int i = 0; i < 8; ++i) {
            int r = rowbase + i + 8 * lhalf;
            int c = nt * 16 + lrow;
            __bf16 h = to_bf16(acc[nt][i]);
            if (wave == 0)      qout[(size_t)r * DD + c]   = h;
            else if (wave == 1) kout[(size_t)r * DD + c]   = h;
            else                vtout[(size_t)c * BSR + r] = h; // transposed V
        }
    }
}

// ---------------------------------------------------------------------------
// Kernel 2: causal flash attention, one wave per 16-row q tile.
// grid = BSR/16/4 blocks, 128 threads (4 waves, independent tiles).
// ---------------------------------------------------------------------------
__global__ __launch_bounds__(128)
void attn_kernel(const __bf16* __restrict__ qw,
                 const __bf16* __restrict__ kw,
                 const __bf16* __restrict__ vt,
                 float* __restrict__ out)
{
    // per-wave P staging: 16 rows x 32 bf16, row stride 72 (conflict-free,
    // 16B-aligned rows)
    __shared__ __align__(16) __bf16 plds[4][16 * 72];

    const int tid   = threadIdx.x;
    const int wave  = tid >> 5;
    const int lane  = tid & 31;
    const int lrow  = lane & 15;
    const int lhalf = lane >> 4;
    const int koff  = 8 * lhalf;

    const int tile  = blockIdx.x * 4 + wave;  // 0..1023
    const int b     = tile >> 7;              // / (S/16)
    const int qi    = tile & 127;
    const int qbase = qi * 16;
    const size_t browbase = (size_t)b * SS;

    __bf16* P = &plds[wave][0];

    // ---- load Q tile as two 16x32 A-fragments (d = 0..31, 32..63)
    Frag16 qa0, qa1;
    {
        const __bf16* qrow = &qw[(browbase + qbase + lrow) * DD];
        qa0.q[0] = *reinterpret_cast<const uint4*>(&qrow[2 * koff]);
        qa0.q[1] = *reinterpret_cast<const uint4*>(&qrow[8 + 2 * koff]);
        qa1.q[0] = *reinterpret_cast<const uint4*>(&qrow[32 + 2 * koff]);
        qa1.q[1] = *reinterpret_cast<const uint4*>(&qrow[40 + 2 * koff]);
    }

    v8f o[4];
    o[0] = v8f{}; o[1] = v8f{}; o[2] = v8f{}; o[3] = v8f{};
    float m[8], l[8];
#pragma unroll
    for (int i = 0; i < 8; ++i) { m[i] = -1e30f; l[i] = 0.0f; }

    const float scale = 0.125f; // D^-0.5, D=64
    const int nkt = (qbase + 15) / 32 + 1; // number of 32-key tiles (causal)

    for (int t = 0; t < nkt; ++t) {
        const int kbase = t * 32;

        if (t + 1 < nkt) { // prefetch next tile (global_prefetch_b8)
            __builtin_prefetch(&kw[(browbase + kbase + 32 + lrow) * DD], 0, 1);
            __builtin_prefetch(&vt[(size_t)lrow * BSR + browbase + kbase + 32], 0, 1);
        }

        // ---- all four K fragments first (one load clause, 8x b128)
        Frag16 k00, k01, k10, k11;   // k<subtile><dfrag>
        {
            const __bf16* kr0 = &kw[(browbase + kbase + lrow) * DD + 16 * lhalf];
            const __bf16* kr1 = &kw[(browbase + kbase + 16 + lrow) * DD + 16 * lhalf];
            k00.q[0] = *reinterpret_cast<const uint4*>(&kr0[0]);
            k00.q[1] = *reinterpret_cast<const uint4*>(&kr0[8]);
            k01.q[0] = *reinterpret_cast<const uint4*>(&kr0[32]);
            k01.q[1] = *reinterpret_cast<const uint4*>(&kr0[40]);
            k10.q[0] = *reinterpret_cast<const uint4*>(&kr1[0]);
            k10.q[1] = *reinterpret_cast<const uint4*>(&kr1[8]);
            k11.q[0] = *reinterpret_cast<const uint4*>(&kr1[32]);
            k11.q[1] = *reinterpret_cast<const uint4*>(&kr1[40]);
        }

        // ---- S = Q @ K^T : two 16x16 C tiles (key subtiles), K-dim = 64
        v8f s0 = v8f{}, s1 = v8f{};
        s0 = wmma_bf16(qa0.v, k00.v, s0);
        s0 = wmma_bf16(qa1.v, k01.v, s0);
        s1 = wmma_bf16(qa0.v, k10.v, s1);
        s1 = wmma_bf16(qa1.v, k11.v, s1);

        // ---- scale + causal mask (only diagonal-straddling tiles)
        const bool needMask = (kbase + 31 > qbase);
        float sv0[8], sv1[8];
#pragma unroll
        for (int i = 0; i < 8; ++i) {
            float a = s0[i] * scale;
            float c = s1[i] * scale;
            if (needMask) {
                int qr = qbase + i + 8 * lhalf;
                if (kbase + lrow > qr)      a = -1e30f;
                if (kbase + 16 + lrow > qr) c = -1e30f;
            }
            sv0[i] = a; sv1[i] = c;
        }

        // ---- online softmax (row reductions across 16-lane halves)
#pragma unroll
        for (int i = 0; i < 8; ++i) {
            float rm = fmaxf(sv0[i], sv1[i]);
            rm = fmaxf(rm, __shfl_xor(rm, 1));
            rm = fmaxf(rm, __shfl_xor(rm, 2));
            rm = fmaxf(rm, __shfl_xor(rm, 4));
            rm = fmaxf(rm, __shfl_xor(rm, 8));
            float newm  = fmaxf(m[i], rm);
            float alpha = __expf(m[i] - newm);
            float p0    = __expf(sv0[i] - newm);
            float p1    = __expf(sv1[i] - newm);
            float rs = p0 + p1;
            rs += __shfl_xor(rs, 1);
            rs += __shfl_xor(rs, 2);
            rs += __shfl_xor(rs, 4);
            rs += __shfl_xor(rs, 8);
            l[i] = l[i] * alpha + rs;
            m[i] = newm;
            o[0][i] *= alpha; o[1][i] *= alpha;
            o[2][i] *= alpha; o[3][i] *= alpha;
            // write P (C layout -> row-major bf16 in LDS)
            int r = i + 8 * lhalf;
            P[r * 72 + lrow]      = to_bf16(p0);
            P[r * 72 + 16 + lrow] = to_bf16(p1);
        }
        // wave-private LDS store->load ordering (no workgroup barrier:
        // waves have different trip counts)
        asm volatile("s_wait_dscnt 0" ::: "memory");

        // ---- repack P as 16x32 A-fragment; V fragments batched (8x b128)
        Frag16 pa;
        pa.q[0] = *reinterpret_cast<const uint4*>(&P[lrow * 72 + 2 * koff]);
        pa.q[1] = *reinterpret_cast<const uint4*>(&P[lrow * 72 + 8 + 2 * koff]);

        Frag16 vb0, vb1, vb2, vb3;
        {
            const size_t vcol = browbase + kbase + 16 * lhalf;
            const __bf16* vp0 = &vt[(size_t)(0 * 16 + lrow) * BSR + vcol];
            const __bf16* vp1 = &vt[(size_t)(1 * 16 + lrow) * BSR + vcol];
            const __bf16* vp2 = &vt[(size_t)(2 * 16 + lrow) * BSR + vcol];
            const __bf16* vp3 = &vt[(size_t)(3 * 16 + lrow) * BSR + vcol];
            vb0.q[0] = *reinterpret_cast<const uint4*>(&vp0[0]);
            vb0.q[1] = *reinterpret_cast<const uint4*>(&vp0[8]);
            vb1.q[0] = *reinterpret_cast<const uint4*>(&vp1[0]);
            vb1.q[1] = *reinterpret_cast<const uint4*>(&vp1[8]);
            vb2.q[0] = *reinterpret_cast<const uint4*>(&vp2[0]);
            vb2.q[1] = *reinterpret_cast<const uint4*>(&vp2[8]);
            vb3.q[0] = *reinterpret_cast<const uint4*>(&vp3[0]);
            vb3.q[1] = *reinterpret_cast<const uint4*>(&vp3[8]);
        }

        // ---- O += P @ V : back-to-back WMMA chain
        o[0] = wmma_bf16(pa.v, vb0.v, o[0]);
        o[1] = wmma_bf16(pa.v, vb1.v, o[1]);
        o[2] = wmma_bf16(pa.v, vb2.v, o[2]);
        o[3] = wmma_bf16(pa.v, vb3.v, o[3]);
    }

    // ---- normalize and store f32 output [B][S][D]
#pragma unroll
    for (int i = 0; i < 8; ++i) {
        float inv = 1.0f / l[i];
        int r = qbase + i + 8 * lhalf;
        float* orow = &out[(browbase + r) * DD];
        orow[0 * 16 + lrow] = o[0][i] * inv;
        orow[1 * 16 + lrow] = o[1][i] * inv;
        orow[2 * 16 + lrow] = o[2][i] * inv;
        orow[3 * 16 + lrow] = o[3][i] * inv;
    }
}

// ---------------------------------------------------------------------------
extern "C" void kernel_launch(void* const* d_in, const int* in_sizes, int n_in,
                              void* d_out, int out_size, void* d_ws, size_t ws_size,
                              hipStream_t stream) {
    const float* x  = (const float*)d_in[0];
    const float* Wq = (const float*)d_in[1];
    const float* Wk = (const float*)d_in[2];
    const float* Wv = (const float*)d_in[3];

    // workspace: q (2MB) | k (2MB) | vT (2MB) | Wt (384KB), all bf16
    __bf16* qw = (__bf16*)d_ws;
    __bf16* kw = qw + (size_t)BSR * DD;
    __bf16* vt = kw + (size_t)BSR * DD;
    __bf16* wt = vt + (size_t)BSR * DD;
    float* out = (float*)d_out;

    wcvt_kernel<<<(3 * DD * EE) / 256, 256, 0, stream>>>(Wq, Wk, Wv, wt);
    qkv_proj_kernel<<<BSR / 16, 96, 0, stream>>>(x, wt, qw, kw, vt);
    attn_kernel<<<BSR / 16 / 4, 128, 0, stream>>>(qw, kw, vt, out);
}
